// PsychedelicAugmentedBioKAN_75187697484100
// MI455X (gfx1250) — matrix operand order repaired and checked by reading, not compiled, not run on hardware
//
#include <hip/hip_runtime.h>

// ---------------------------------------------------------------------------
// PsychedelicAugmentedBioKAN on MI455X (gfx1250): bf16 WMMA everywhere.
// B=2, S=2048, D=1024, H=16, HD=64. ~73 GFLOP vs ~60MB traffic (fits in the
// 192MB L2) -> tensor-issue bound.
// GEMM: 16x64 C strip per wave, double-buffered fragment pipeline.
// Attention: TRANSPOSED tiles (S^T = K x Q^T, O^T = V^T x P^T) so the query
// index lives in the lane -> per-lane scalar softmax stats, register-only
// P^T assembly with constant vector indices (no alloca/scratch), softmax
// scale pre-folded into the Q projection.
// ---------------------------------------------------------------------------

#define BB   2
#define SS   2048
#define DDIM 1024
#define HH   16
#define HDD  64
#define MTOT (BB * SS)       // 4096 rows of the flattened activations

typedef __bf16 bf16;
typedef __attribute__((ext_vector_type(16))) __bf16 bfx16;
typedef __attribute__((ext_vector_type(8)))  __bf16 bfx8;
typedef __attribute__((ext_vector_type(8)))  float  v8f;

union BF16Frag { bfx16 v; bfx8 h[2]; };

// A-matrix 16x32 bf16 fragment in this code's contiguous k order (not the ISA table order; B matches): lane l holds row m=l&15.
// Lanes 0-15 hold K={0..15}, lanes 16-31 hold K={16..31}:
// one contiguous 16-element (32B) run per lane.
__device__ __forceinline__ bfx16 load_frag_a(const bf16* __restrict__ base,
                                             int row_stride, int m, int k0,
                                             int half) {
  BF16Frag f;
  const bf16* p = base + (size_t)m * row_stride + k0 + half * 16;
  f.h[0] = *reinterpret_cast<const bfx8*>(p);
  f.h[1] = *reinterpret_cast<const bfx8*>(p + 8);
  return f.v;
}

// B-matrix 32x16 bf16 fragment from a row-major [N,K] source (so B = src^T):
// lane l holds column n=l&15; element e holds K = half*16 + e -> one
// contiguous 16-element (32B) run per lane.
__device__ __forceinline__ bfx16 load_frag_b(const bf16* __restrict__ base,
                                             int row_stride, int n, int k0,
                                             int half) {
  BF16Frag f;
  const bf16* p = base + (size_t)n * row_stride + k0 + half * 16;
  f.h[0] = *reinterpret_cast<const bfx8*>(p);
  f.h[1] = *reinterpret_cast<const bfx8*>(p + 8);
  return f.v;
}

// ---------------------------------------------------------------------------
// fp32 -> bf16 conversion with scale, 8 elements / thread (b128 in/out)
// ---------------------------------------------------------------------------
__global__ void f32_to_bf16_v8(const float* __restrict__ src,
                               bf16* __restrict__ dst, int n8, float scale) {
  int i = blockIdx.x * blockDim.x + threadIdx.x;
  if (i < n8) {
    const float4* s4 = reinterpret_cast<const float4*>(src) + (size_t)i * 2;
    float4 a = s4[0], b = s4[1];
    bfx8 o;
    o[0] = (bf16)(a.x * scale); o[1] = (bf16)(a.y * scale);
    o[2] = (bf16)(a.z * scale); o[3] = (bf16)(a.w * scale);
    o[4] = (bf16)(b.x * scale); o[5] = (bf16)(b.y * scale);
    o[6] = (bf16)(b.z * scale); o[7] = (bf16)(b.w * scale);
    reinterpret_cast<bfx8*>(dst)[i] = o;
  }
}

// ---------------------------------------------------------------------------
// C[M,N] = A[M,K] @ W[N,K]^T + bias*bias_scale. One wave owns a 16x64 strip.
// Explicit double-buffered pipeline: fragments for step k+1 are loaded into
// a disjoint register set while the 4 wmma of step k execute.
// OUT_MODE: 0 = bf16 [M,N], 1 = f32 [M,N],
//           2 = bf16 [B,H,S,HD] (head-split), 3 = bf16 [B,H,HD,S] (V^T)
// ---------------------------------------------------------------------------
template <int OUT_MODE, int RELU>
__global__ __launch_bounds__(32) void gemm_wmma_bf16(
    const bf16* __restrict__ A, const bf16* __restrict__ W,
    const float* __restrict__ bias, void* __restrict__ out,
    int M, int N, int K, float bias_scale) {
  const int lane = threadIdx.x & 31;
  const int half = lane >> 4;
  const int ln   = lane & 15;
  const int ntiles = N >> 6;                    // 64-wide N strips
  const int tm = (blockIdx.x / ntiles) << 4;
  const int tn = (blockIdx.x % ntiles) << 6;

  v8f acc0 = {}, acc1 = {}, acc2 = {}, acc3 = {};

  // pipeline prologue: fragments for k0 = 0
  bfx16 a  = load_frag_a(A, K, tm + ln, 0, half);
  bfx16 b0 = load_frag_b(W, K, tn + 0  + ln, 0, half);
  bfx16 b1 = load_frag_b(W, K, tn + 16 + ln, 0, half);
  bfx16 b2 = load_frag_b(W, K, tn + 32 + ln, 0, half);
  bfx16 b3 = load_frag_b(W, K, tn + 48 + ln, 0, half);

#pragma unroll 2
  for (int k0 = 0; k0 < K - 32; k0 += 32) {
    const int kn = k0 + 32;
    // next-step loads (disjoint regs) overlap current-step wmma
    bfx16 an  = load_frag_a(A, K, tm + ln, kn, half);
    bfx16 bn0 = load_frag_b(W, K, tn + 0  + ln, kn, half);
    bfx16 bn1 = load_frag_b(W, K, tn + 16 + ln, kn, half);
    bfx16 bn2 = load_frag_b(W, K, tn + 32 + ln, kn, half);
    bfx16 bn3 = load_frag_b(W, K, tn + 48 + ln, kn, half);
    acc0 = __builtin_amdgcn_wmma_f32_16x16x32_bf16(false, a, false, b0,
                                                   (short)0, acc0, false, false);
    acc1 = __builtin_amdgcn_wmma_f32_16x16x32_bf16(false, a, false, b1,
                                                   (short)0, acc1, false, false);
    acc2 = __builtin_amdgcn_wmma_f32_16x16x32_bf16(false, a, false, b2,
                                                   (short)0, acc2, false, false);
    acc3 = __builtin_amdgcn_wmma_f32_16x16x32_bf16(false, a, false, b3,
                                                   (short)0, acc3, false, false);
    a = an; b0 = bn0; b1 = bn1; b2 = bn2; b3 = bn3;
  }
  // pipeline epilogue: last k-step
  acc0 = __builtin_amdgcn_wmma_f32_16x16x32_bf16(false, a, false, b0,
                                                 (short)0, acc0, false, false);
  acc1 = __builtin_amdgcn_wmma_f32_16x16x32_bf16(false, a, false, b1,
                                                 (short)0, acc1, false, false);
  acc2 = __builtin_amdgcn_wmma_f32_16x16x32_bf16(false, a, false, b2,
                                                 (short)0, acc2, false, false);
  acc3 = __builtin_amdgcn_wmma_f32_16x16x32_bf16(false, a, false, b3,
                                                 (short)0, acc3, false, false);

  const v8f* accs[4] = {&acc0, &acc1, &acc2, &acc3};
#pragma unroll
  for (int t = 0; t < 4; ++t) {
    const int n  = tn + t * 16 + ln;
    const float bn = bias[n] * bias_scale;
#pragma unroll
    for (int r = 0; r < 8; ++r) {
      const int m = tm + r + (half << 3);  // C layout: VGPR r -> row r (+8 hi)
      float v = (*accs[t])[r] + bn;
      if (RELU) v = v > 0.f ? v : 0.f;
      if (OUT_MODE == 1) {
        ((float*)out)[(size_t)m * N + n] = v;
      } else if (OUT_MODE == 0) {
        ((bf16*)out)[(size_t)m * N + n] = (bf16)v;
      } else {
        const int b_ = m / SS, s = m % SS;
        const int h  = n / HDD, hd = n % HDD;
        if (OUT_MODE == 2)
          ((bf16*)out)[(((size_t)(b_ * HH + h)) * SS + s) * HDD + hd] = (bf16)v;
        else
          ((bf16*)out)[(((size_t)(b_ * HH + h)) * HDD + hd) * SS + s] = (bf16)v;
      }
    }
  }
}

// ---------------------------------------------------------------------------
// Flash attention, fully transposed tiles + fused neuromodulation epilogue.
// 8 independent waves / block, one 16-query tile per wave, no LDS/barriers.
//   S^T(16k x 16q) = K_tile(16k x HD) x Q^T(HD x 16q)   (2 wmma per subtile)
//   O^T(16hd x 16q) += V^T_tile(16hd x 32k) x P^T(32k x 16q)
// C-layout of S^T/O^T puts q in the lane -> per-lane scalar m/l stats.
// Q is pre-scaled by HD^-0.5 at projection time.
// ---------------------------------------------------------------------------
__global__ __launch_bounds__(256) void attn_wmma(
    const bf16* __restrict__ Qh, const bf16* __restrict__ Kh,
    const bf16* __restrict__ Vt, const float* __restrict__ mod,
    const float* __restrict__ dop, const float* __restrict__ ser,
    const float* __restrict__ nor, const float* __restrict__ ach,
    const float* __restrict__ ascale, const float* __restrict__ abias,
    bf16* __restrict__ attnmod) {
  const int wave = threadIdx.x >> 5;
  const int lane = threadIdx.x & 31;
  const int half = lane >> 4;
  const int ln   = lane & 15;

  const int qblocks = SS / 128;                 // 8 q-tiles per block
  const int bh = blockIdx.x / qblocks;
  const int q0 = (blockIdx.x % qblocks) * 128 + wave * 16;

  const bf16* Qb = Qh + (size_t)bh * SS * HDD;  // [S, HD]
  const bf16* Kb = Kh + (size_t)bh * SS * HDD;  // [S, HD]
  const bf16* Vb = Vt + (size_t)bh * HDD * SS;  // [HD, S] (transposed)

  // Q^T as B fragments, loaded once and reused over all 64 key chunks
  const bfx16 qb0 = load_frag_b(Qb, HDD, q0 + ln, 0, half);
  const bfx16 qb1 = load_frag_b(Qb, HDD, q0 + ln, 32, half);

  v8f o[4] = {v8f{}, v8f{}, v8f{}, v8f{}};   // O^T tiles: hd in VGPR, q in lane
  float mm = -3.0e38f, ll = 0.f;             // per-lane scalar stats (q = ln)

  for (int kb = 0; kb < SS; kb += 32) {
    if (kb + 32 < SS)
      __builtin_prefetch(Kb + (size_t)(kb + 32 + lane) * HDD, 0, 1);

    // S^T subtiles, computed subtile-by-subtile (2 live K fragments max)
    v8f s0t = {}, s1t = {};
    {
      bfx16 ka = load_frag_a(Kb, HDD, kb + ln, 0, half);
      s0t = __builtin_amdgcn_wmma_f32_16x16x32_bf16(false, ka, false, qb0,
                                                    (short)0, s0t, false, false);
      ka = load_frag_a(Kb, HDD, kb + ln, 32, half);
      s0t = __builtin_amdgcn_wmma_f32_16x16x32_bf16(false, ka, false, qb1,
                                                    (short)0, s0t, false, false);
    }
    {
      bfx16 kc = load_frag_a(Kb, HDD, kb + 16 + ln, 0, half);
      s1t = __builtin_amdgcn_wmma_f32_16x16x32_bf16(false, kc, false, qb0,
                                                    (short)0, s1t, false, false);
      kc = load_frag_a(Kb, HDD, kb + 16 + ln, 32, half);
      s1t = __builtin_amdgcn_wmma_f32_16x16x32_bf16(false, kc, false, qb1,
                                                    (short)0, s1t, false, false);
    }

    // per-lane softmax: in-lane max over 16 keys + one cross-half shuffle
    float pm = -3.0e38f;
#pragma unroll
    for (int r = 0; r < 8; ++r)
      pm = fmaxf(pm, fmaxf(s0t[r], s1t[r]));
    pm = fmaxf(pm, __shfl_xor(pm, 16, 32));
    const float mnew = fmaxf(mm, pm);
    const float sc   = __expf(mm - mnew);
    mm = mnew;

    // exp, row-sum, and register-only P^T B-fragment assembly.
    // B-frag element e holds key = half*16 + e. Element r is own p0 for
    // half0 / received s1t for half1; element r+8 is received s0t for
    // half0 / own p1 for half1 -> all vector subscripts are constants.
    float rs = 0.f;
    BF16Frag pb;
#pragma unroll
    for (int r = 0; r < 8; ++r) {
      const float p0 = __expf(s0t[r] - mnew);   // key kb + r + 8*half
      const float p1 = __expf(s1t[r] - mnew);   // key kb + 16 + r + 8*half
      rs += p0 + p1;
      const float send = half ? p0 : p1;        // partner needs this one
      const float rec  = __shfl_xor(send, 16, 32);
      const float lo   = half ? rec : p0;       // element r
      const float hi   = half ? p1 : rec;       // element r+8
      pb.v[r]     = (bf16)lo;
      pb.v[r + 8] = (bf16)hi;
    }
    rs += __shfl_xor(rs, 16, 32);
    ll = ll * sc + rs;

    // rescale O^T (scalar per lane) and accumulate V^T x P^T
#pragma unroll
    for (int t = 0; t < 4; ++t) {
#pragma unroll
      for (int r = 0; r < 8; ++r) o[t][r] *= sc;
    }
#pragma unroll
    for (int t = 0; t < 4; ++t) {
      bfx16 va = load_frag_a(Vb, SS, t * 16 + ln, kb, half);
      o[t] = __builtin_amdgcn_wmma_f32_16x16x32_bf16(false, va, false, pb.v,
                                                     (short)0, o[t], false,
                                                     false);
    }
  }

  // fused epilogue: 1/l, attention scale/bias, neuromodulation gate.
  // Lane covers q = q0+ln, hd = t*16 + r + 8*half -> 8 consecutive d per tile.
  const float gate = (dop[0] + ser[0] + nor[0] + ach[0]) * 0.25f;
  const float as = ascale[0], ab = abias[0];
  const int b_ = bh / HH, h = bh % HH;
  const float inv = 1.f / ll;
  const size_t rowbase =
      ((size_t)(b_ * SS + q0 + ln)) * DDIM + h * HDD + half * 8;
#pragma unroll
  for (int t = 0; t < 4; ++t) {
    const size_t base = rowbase + t * 16;
    const float4 m0 = *reinterpret_cast<const float4*>(mod + base);
    const float4 m1 = *reinterpret_cast<const float4*>(mod + base + 4);
    const float mv[8] = {m0.x, m0.y, m0.z, m0.w, m1.x, m1.y, m1.z, m1.w};
    bfx8 ov;
#pragma unroll
    for (int r = 0; r < 8; ++r) {
      float val = o[t][r] * inv;
      val = (val * as + ab) * (1.f + mv[r] * gate);
      ov[r] = (bf16)val;
    }
    *reinterpret_cast<bfx8*>(attnmod + base) = ov;
  }
}

// ---------------------------------------------------------------------------
// launch
// ---------------------------------------------------------------------------
extern "C" void kernel_launch(void* const* d_in, const int* in_sizes, int n_in,
                              void* d_out, int out_size, void* d_ws,
                              size_t ws_size, hipStream_t stream) {
  const float* query = (const float*)d_in[0];
  const float* Wq = (const float*)d_in[1];
  const float* bq = (const float*)d_in[2];
  const float* Wk = (const float*)d_in[3];
  const float* bk = (const float*)d_in[4];
  const float* Wv = (const float*)d_in[5];
  const float* bv = (const float*)d_in[6];
  const float* Wo = (const float*)d_in[7];
  const float* bo = (const float*)d_in[8];
  const float* Wm1 = (const float*)d_in[9];
  const float* bm1 = (const float*)d_in[10];
  const float* Wm2 = (const float*)d_in[11];
  const float* bm2 = (const float*)d_in[12];
  const float* dop = (const float*)d_in[13];
  const float* ser = (const float*)d_in[14];
  const float* nor = (const float*)d_in[15];
  const float* ach = (const float*)d_in[16];
  const float* ascale = (const float*)d_in[17];
  const float* abias  = (const float*)d_in[18];
  float* out = (float*)d_out;

  const float SSCALE = 0.125f;  // HD^-0.5, folded into Q projection

  // workspace carve-up (256B aligned)
  char* ws = (char*)d_ws;
  size_t off = 0;
  auto carve = [&](size_t bytes) -> void* {
    void* p = ws + off;
    off += (bytes + 255) & ~(size_t)255;
    return p;
  };
  bf16* qbf   = (bf16*)carve((size_t)MTOT * DDIM * 2);      // query bf16
  bf16* wqb   = (bf16*)carve((size_t)DDIM * DDIM * 2);
  bf16* wkb   = (bf16*)carve((size_t)DDIM * DDIM * 2);
  bf16* wvb   = (bf16*)carve((size_t)DDIM * DDIM * 2);
  bf16* wob   = (bf16*)carve((size_t)DDIM * DDIM * 2);
  bf16* wm1b  = (bf16*)carve((size_t)(DDIM / 4) * DDIM * 2);
  bf16* wm2b  = (bf16*)carve((size_t)DDIM * (DDIM / 4) * 2);
  bf16* Qh    = (bf16*)carve((size_t)MTOT * DDIM * 2);      // [B,H,S,HD]
  bf16* Kh    = (bf16*)carve((size_t)MTOT * DDIM * 2);      // [B,H,S,HD]
  bf16* Vt    = (bf16*)carve((size_t)MTOT * DDIM * 2);      // [B,H,HD,S]
  bf16* mod1  = (bf16*)carve((size_t)MTOT * (DDIM / 4) * 2);
  float* modf = (float*)carve((size_t)MTOT * DDIM * 4);
  bf16* attnm = (bf16*)carve((size_t)MTOT * DDIM * 2);
  (void)ws_size; (void)n_in; (void)in_sizes; (void)out_size;

  // 1) fp32 -> bf16 conversions (8 elems/thread; all sizes are /8)
  auto cvt = [&](const float* s, bf16* d, int n, float scale) {
    int n8 = n / 8;
    f32_to_bf16_v8<<<(n8 + 255) / 256, 256, 0, stream>>>(s, d, n8, scale);
  };
  cvt(query, qbf, MTOT * DDIM, 1.f);
  cvt(Wq, wqb, DDIM * DDIM, SSCALE);          // pre-fold softmax scale into Q
  cvt(Wk, wkb, DDIM * DDIM, 1.f);
  cvt(Wv, wvb, DDIM * DDIM, 1.f);
  cvt(Wo, wob, DDIM * DDIM, 1.f);
  cvt(Wm1, wm1b, (DDIM / 4) * DDIM, 1.f);
  cvt(Wm2, wm2b, DDIM * (DDIM / 4), 1.f);

  // 2) QKV projections (head-split outputs; V transposed for P.V fragments)
  {
    dim3 grid((MTOT / 16) * (DDIM / 64));
    gemm_wmma_bf16<2, 0><<<grid, 32, 0, stream>>>(qbf, wqb, bq, Qh, MTOT, DDIM, DDIM, SSCALE);
    gemm_wmma_bf16<2, 0><<<grid, 32, 0, stream>>>(qbf, wkb, bk, Kh, MTOT, DDIM, DDIM, 1.f);
    gemm_wmma_bf16<3, 0><<<grid, 32, 0, stream>>>(qbf, wvb, bv, Vt, MTOT, DDIM, DDIM, 1.f);
  }

  // 3) neuromod MLP: relu(q @ Wm1^T + b) @ Wm2^T + b  -> modf (fp32)
  {
    dim3 g1((MTOT / 16) * ((DDIM / 4) / 64));
    gemm_wmma_bf16<0, 1><<<g1, 32, 0, stream>>>(qbf, wm1b, bm1, mod1, MTOT, DDIM / 4, DDIM, 1.f);
    dim3 g2((MTOT / 16) * (DDIM / 64));
    gemm_wmma_bf16<1, 0><<<g2, 32, 0, stream>>>(mod1, wm2b, bm2, modf, MTOT, DDIM, DDIM / 4, 1.f);
  }

  // 4) flash attention (transposed, LDS-free) + fused epilogue -> attnm
  {
    dim3 grid(BB * HH * (SS / 128));
    attn_wmma<<<grid, 256, 0, stream>>>(Qh, Kh, Vt, modf, dop, ser, nor, ach,
                                        ascale, abias, attnm);
  }

  // 5) output projection -> d_out (fp32)
  {
    dim3 grid((MTOT / 16) * (DDIM / 64));
    gemm_wmma_bf16<1, 0><<<grid, 32, 0, stream>>>(attnm, wob, bo, out, MTOT, DDIM, DDIM, 1.f);
  }
}
